// MambaPretrain_33105607917974
// MI455X (gfx1250) — matrix operand order repaired and checked by reading, not compiled, not run on hardware
//
#include <hip/hip_runtime.h>
#include <math.h>

#define B_    4
#define L_    2500
#define NL_   12
#define DM_   512
#define DS_   16
#define DI_   1024
#define DC_   4
#define DTR_  32
#define KEEP_ 1250

typedef __attribute__((ext_vector_type(16))) __bf16 v16bf;
typedef __attribute__((ext_vector_type(8)))  float  v8f;
typedef int v4i_ __attribute__((vector_size(16)));

#define AS1_ __attribute__((address_space(1)))
#define AS3_ __attribute__((address_space(3)))

#if defined(__HIP_DEVICE_COMPILE__) && \
    __has_builtin(__builtin_amdgcn_global_load_async_to_lds_b128) && \
    __has_builtin(__builtin_amdgcn_s_wait_asynccnt)
#define ASYNC_OK 1
#else
#define ASYNC_OK 0
#endif

__device__ __forceinline__ float silu_(float x){ return x / (1.f + __expf(-x)); }
__device__ __forceinline__ float softplus_(float x){ return x > 30.f ? x : log1pf(__expf(x)); }

// 16-byte global -> LDS stage (async on CDNA5, sync fallback otherwise)
__device__ __forceinline__ void stage16_(const __bf16* g, __bf16* l) {
#if ASYNC_OK
  __builtin_amdgcn_global_load_async_to_lds_b128((AS1_ v4i_*)g, (AS3_ v4i_*)l, 0, 0);
#else
  *reinterpret_cast<uint4*>(l) = *reinterpret_cast<const uint4*>(g);
#endif
}
__device__ __forceinline__ void stage_wait_() {
#if ASYNC_OK
  __builtin_amdgcn_s_wait_asynccnt(0);
#endif
}

// ---------------------------------------------------------- f32 -> bf16 -----
__global__ void cvt_bf16_kernel(const float* __restrict__ src, __bf16* __restrict__ dst, long n) {
  long i = (long)blockIdx.x * blockDim.x + threadIdx.x;
  if (i < n) dst[i] = (__bf16)src[i];
}

// ---------------------------------------------------------------- masking ---
// rank[i] = #{j : n[j] < n[i] || (n[j]==n[i] && j<i)}  == ids_restore[i]
__global__ void rank_kernel(const float* __restrict__ noise,
                            int* ids_shuffle, int* ids_restore, float* mask) {
  int b = blockIdx.y;
  int i = blockIdx.x * blockDim.x + threadIdx.x;
  if (i >= L_) return;
  const float* row = noise + b * L_;
  float v = row[i];
  int cnt = 0;
  for (int j = 0; j < L_; ++j) {
    float w = row[j];
    cnt += (w < v) || (w == v && j < i);
  }
  ids_restore[b * L_ + i] = cnt;
  ids_shuffle[b * L_ + cnt] = i;
  mask[b * L_ + i] = (cnt < KEEP_) ? 0.f : 1.f;
}

// ------------------------------------------------------------------ embed ---
__global__ void embed_kernel(const float* __restrict__ ecg, const int* __restrict__ ids_shuffle,
                             const float* __restrict__ W_in, const float* __restrict__ b_in,
                             float* __restrict__ X) {
  int idx = blockIdx.x * blockDim.x + threadIdx.x;
  if (idx >= B_ * KEEP_ * DM_) return;
  int c = idx % DM_;
  int r = idx / DM_;
  int b = r / KEEP_;
  int k = r % KEEP_;
  int id = ids_shuffle[b * L_ + k];
  float acc = b_in[c];
#pragma unroll
  for (int j = 0; j < NL_; ++j)
    acc += ecg[(b * NL_ + j) * L_ + id] * W_in[c * NL_ + j];
  X[idx] = acc;
}

// ----------------------------------------------------- WMMA bf16 GEMM -------
// C[M,N] = A[M,K](lda,bf16) * W[N,K](ldw,bf16)^T (+bias[n]) (act1=softplus)
// Block: 128 thr / 4 waves -> 32x64 C macro-tile. Per K-step(32): async-stage
// A 32x32 + B 64x32 bf16 tiles to LDS, then 2 WMMAs per wave (A frag reused).
__global__ void gemm_wmma_bf16(const __bf16* __restrict__ A, int lda,
                               const __bf16* __restrict__ W, int ldw,
                               float* __restrict__ C, int ldc,
                               int M, int N, int K,
                               const float* __restrict__ bias, int act) {
  __shared__ __bf16 As[32 * 32];   // [m][k] row-major (2 KB)
  __shared__ __bf16 Bs[64 * 32];   // [n][k] row-major (4 KB)
  int t    = threadIdx.x;
  int lane = t & 31;
  int wid  = t >> 5;
  int waveM = wid & 1;             // 2 waves over M (16 rows each)
  int waveN = wid >> 1;            // 2 waves over N (32 cols each)
  int hf  = lane >> 4;             // lane half
  int l15 = lane & 15;
  int mBase = blockIdx.y << 5;
  int nBase = blockIdx.x << 6;

  int aRow = t >> 2, aCh = t & 3;  // A stage: 128 chunks of 16B
  int gm = mBase + aRow; if (gm >= M) gm = M - 1;
  const __bf16* gA = A + (size_t)gm * lda + aCh * 8;

  v8f acc[2];
#pragma unroll
  for (int f = 0; f < 2; ++f) acc[f] = (v8f){0.f,0.f,0.f,0.f,0.f,0.f,0.f,0.f};

  for (int k0 = 0; k0 < K; k0 += 32) {
    // ---- stage tiles (3 x b128 per thread) ----
    stage16_(gA + k0, &As[aRow * 32 + aCh * 8]);
#pragma unroll
    for (int i = 0; i < 2; ++i) {
      int idx = t + i * 128;                 // B stage: 256 chunks of 16B
      int bRow = idx >> 2, bCh = idx & 3;
      stage16_(W + (size_t)(nBase + bRow) * ldw + k0 + bCh * 8,
               &Bs[bRow * 32 + bCh * 8]);
    }
    stage_wait_();
    __syncthreads();

    // ---- A fragment: m = lane&15 ; k = 16*(p>>2) + 8*half + 2*(p&3) + h ----
    v16bf a;
    const __bf16* ap = &As[(waveM * 16 + l15) * 32];
#pragma unroll
    for (int p = 0; p < 8; ++p) {
      int k = ((p >> 2) << 4) + (hf << 3) + ((p & 3) << 1);
      a[2 * p]     = ap[k];
      a[2 * p + 1] = ap[k + 1];
    }
    // ---- two N fragments share the A fragment ----
#pragma unroll
    for (int f = 0; f < 2; ++f) {
      v16bf bfrag;
      const __bf16* bp = &Bs[(waveN * 32 + f * 16 + l15) * 32];
#pragma unroll
      for (int p = 0; p < 8; ++p) {          // n = lane&15 ; k = 16*(p>>2) + 8*half + 2*(p&3) + h
        int k = ((p >> 2) << 4) + (hf << 3) + ((p & 3) << 1);
        bfrag[2 * p]     = bp[k];
        bfrag[2 * p + 1] = bp[k + 1];
      }
      acc[f] = __builtin_amdgcn_wmma_f32_16x16x32_bf16(false, a, false, bfrag,
                                                       (short)0, acc[f], false, false);
    }
    __syncthreads();
  }

#pragma unroll
  for (int f = 0; f < 2; ++f) {
    int n = nBase + waveN * 32 + f * 16 + l15;
    float bv = bias ? bias[n] : 0.f;
#pragma unroll
    for (int v = 0; v < 8; ++v) {            // C: m = v + 8*half, n = lane&15
      int m = mBase + waveM * 16 + v + (hf << 3);
      if (m < M) {
        float val = acc[f][v] + bv;
        if (act == 1) val = softplus_(val);
        C[(size_t)m * ldc + n] = val;
      }
    }
  }
}

// ------------------------------------------------ depthwise conv + SiLU -----
__global__ void conv_silu_kernel(const float* __restrict__ inproj,
                                 const float* __restrict__ conv_w,
                                 const float* __restrict__ conv_b,
                                 float* __restrict__ XC, int Lr) {
  long idx = (long)blockIdx.x * blockDim.x + threadIdx.x;
  long total = (long)B_ * Lr * DI_;
  if (idx >= total) return;
  int d  = idx % DI_;
  long bl = idx / DI_;
  int l = bl % Lr;
  int b = bl / Lr;
  float acc = conv_b[d];
#pragma unroll
  for (int j = 0; j < DC_; ++j) {
    int ls = l - (DC_ - 1) + j;
    float xv = (ls >= 0) ? inproj[(long)(b * Lr + ls) * (2 * DI_) + d] : 0.f;
    acc += conv_w[d * DC_ + j] * xv;
  }
  XC[idx] = silu_(acc);
}

// ------------------------------------------------------- selective scan -----
__global__ void scan_kernel(const float* __restrict__ DT, const float* __restrict__ XC,
                            const float* __restrict__ DBC, const float* __restrict__ A_log,
                            const float* __restrict__ Dp, float* __restrict__ Y, int Lr) {
  __shared__ float sB[DS_], sC[DS_];
  int tid = threadIdx.x;
  int b  = blockIdx.x / (DI_ / 256);
  int dg = blockIdx.x % (DI_ / 256);
  int d  = dg * 256 + tid;
  float Av[DS_], h[DS_];
#pragma unroll
  for (int s = 0; s < DS_; ++s) { Av[s] = -__expf(A_log[d * DS_ + s]); h[s] = 0.f; }
  float Dv = Dp[d];
  for (int l = 0; l < Lr; ++l) {
    long row = (long)b * Lr + l;
    if (tid < 2 * DS_) {
      float v = DBC[row * 64 + DTR_ + tid];
      if (tid < DS_) sB[tid] = v; else sC[tid - DS_] = v;
    }
    __syncthreads();
    float dt  = DT[row * DI_ + d];
    float xv  = XC[row * DI_ + d];
    float dtx = dt * xv;
    float y = 0.f;
#pragma unroll
    for (int s = 0; s < DS_; ++s) {
      h[s] = __expf(dt * Av[s]) * h[s] + dtx * sB[s];
      y += h[s] * sC[s];
    }
    Y[row * DI_ + d] = y + xv * Dv;
    __syncthreads();
  }
}

// ------------------------------------------------------------- gating -------
__global__ void gate_kernel(const float* __restrict__ inproj, float* __restrict__ U, long total) {
  long idx = (long)blockIdx.x * blockDim.x + threadIdx.x;
  if (idx >= total) return;
  int d = idx % DI_;
  long row = idx / DI_;
  float z = inproj[row * (2 * DI_) + DI_ + d];
  U[idx] = U[idx] * silu_(z);
}

// ---------------------------------------------------------- layernorm -------
__global__ void layernorm_kernel(const float* __restrict__ Xin, const float* __restrict__ g,
                                 const float* __restrict__ beta, float* __restrict__ Xout) {
  __shared__ float red[256];
  int r = blockIdx.x, tid = threadIdx.x;
  const float* row = Xin + (long)r * DM_;
  float s = 0.f;
  for (int c = tid; c < DM_; c += 256) s += row[c];
  red[tid] = s; __syncthreads();
  for (int o = 128; o > 0; o >>= 1) { if (tid < o) red[tid] += red[tid + o]; __syncthreads(); }
  float mu = red[0] / DM_;
  __syncthreads();
  float v = 0.f;
  for (int c = tid; c < DM_; c += 256) { float tv = row[c] - mu; v += tv * tv; }
  red[tid] = v; __syncthreads();
  for (int o = 128; o > 0; o >>= 1) { if (tid < o) red[tid] += red[tid + o]; __syncthreads(); }
  float rstd = rsqrtf(red[0] / DM_ + 1e-5f);
  for (int c = tid; c < DM_; c += 256)
    Xout[(long)r * DM_ + c] = (row[c] - mu) * rstd * g[c] + beta[c];
}

// -------------------------------------------------- decoder input build -----
__global__ void build_dec_kernel(const float* __restrict__ Xenc, const int* __restrict__ ids_restore,
                                 const float* __restrict__ mask_token, float* __restrict__ Xdec) {
  long idx = (long)blockIdx.x * blockDim.x + threadIdx.x;
  long total = (long)B_ * L_ * DM_;
  if (idx >= total) return;
  int c = idx % DM_;
  long bl = idx / DM_;
  int i = bl % L_;
  int b = bl / L_;
  int r = ids_restore[b * L_ + i];
  Xdec[idx] = (r < KEEP_) ? Xenc[((long)b * KEEP_ + r) * DM_ + c] : mask_token[c];
}

// --------------------------------------------------------------- head -------
__global__ void head_kernel(const float* __restrict__ Xf, const float* __restrict__ W_head,
                            const float* __restrict__ b_head, float* __restrict__ recon) {
  int idx = blockIdx.x * blockDim.x + threadIdx.x;
  if (idx >= B_ * NL_ * L_) return;
  int l = idx % L_;
  int bc = idx / L_;
  int c = bc % NL_;
  int b = bc / NL_;
  const float* row = Xf + ((long)b * L_ + l) * DM_;
  const float* wr  = W_head + c * DM_;
  float acc = b_head[c];
#pragma unroll 4
  for (int k = 0; k < DM_; ++k) acc += row[k] * wr[k];
  recon[idx] = acc;
}

// --------------------------------------------------------------- loss -------
__global__ void err_kernel(const float* __restrict__ recon, const float* __restrict__ ecg,
                           const float* __restrict__ mask, float* __restrict__ errws) {
  int idx = blockIdx.x * blockDim.x + threadIdx.x;
  if (idx >= B_ * L_) return;
  int l = idx % L_;
  int b = idx / L_;
  float s = 0.f;
#pragma unroll
  for (int c = 0; c < NL_; ++c) {
    float dlt = recon[(b * NL_ + c) * L_ + l] - ecg[(b * NL_ + c) * L_ + l];
    s += dlt * dlt;
  }
  errws[idx] = (s / NL_) * mask[idx];
}

__global__ void loss_kernel(const float* __restrict__ errws, const float* __restrict__ mask,
                            float* __restrict__ out) {
  __shared__ float se[256], sm[256];
  int tid = threadIdx.x;
  float e = 0.f, m = 0.f;
  for (int i = tid; i < B_ * L_; i += 256) { e += errws[i]; m += mask[i]; }
  se[tid] = e; sm[tid] = m; __syncthreads();
  for (int o = 128; o > 0; o >>= 1) {
    if (tid < o) { se[tid] += se[tid + o]; sm[tid] += sm[tid + o]; }
    __syncthreads();
  }
  if (tid == 0) out[0] = se[0] / sm[0];
}

// ------------------------------------------------------------- launcher -----
extern "C" void kernel_launch(void* const* d_in, const int* in_sizes, int n_in,
                              void* d_out, int out_size, void* d_ws, size_t ws_size,
                              hipStream_t stream) {
  (void)in_sizes; (void)n_in; (void)out_size; (void)ws_size;
  const float* ecg        = (const float*)d_in[0];
  const float* noise      = (const float*)d_in[1];
  const float* W_in       = (const float*)d_in[2];
  const float* b_in       = (const float*)d_in[3];
  const float* mask_token = (const float*)d_in[4];
  const float* enc_in_w   = (const float*)d_in[5];
  const float* enc_conv_w = (const float*)d_in[6];
  const float* enc_conv_b = (const float*)d_in[7];
  const float* enc_xproj  = (const float*)d_in[8];
  const float* enc_dt_w   = (const float*)d_in[9];
  const float* enc_dt_b   = (const float*)d_in[10];
  const float* enc_A_log  = (const float*)d_in[11];
  const float* enc_D      = (const float*)d_in[12];
  const float* enc_out_w  = (const float*)d_in[13];
  const float* enc_ln_g   = (const float*)d_in[14];
  const float* enc_ln_b   = (const float*)d_in[15];
  const float* dec_in_w   = (const float*)d_in[16];
  const float* dec_conv_w = (const float*)d_in[17];
  const float* dec_conv_b = (const float*)d_in[18];
  const float* dec_xproj  = (const float*)d_in[19];
  const float* dec_dt_w   = (const float*)d_in[20];
  const float* dec_dt_b   = (const float*)d_in[21];
  const float* dec_A_log  = (const float*)d_in[22];
  const float* dec_D      = (const float*)d_in[23];
  const float* dec_out_w  = (const float*)d_in[24];
  const float* dec_ln_g   = (const float*)d_in[25];
  const float* dec_ln_b   = (const float*)d_in[26];
  const float* W_head     = (const float*)d_in[27];
  const float* b_head     = (const float*)d_in[28];

  char* wsb = (char*)d_ws;
  size_t off = 0;
  auto carve = [&](size_t bytes) -> char* {
    char* p = wsb + off;
    off = (off + bytes + 255) & ~(size_t)255;
    return p;
  };
  int*    ids_shuffle = (int*)carve((size_t)B_ * L_ * 4);
  int*    ids_restore = (int*)carve((size_t)B_ * L_ * 4);
  float*  mask        = (float*)carve((size_t)B_ * L_ * 4);
  float*  X           = (float*)carve((size_t)B_ * L_ * DM_ * 4);
  float*  X2          = (float*)carve((size_t)B_ * L_ * DM_ * 4);
  float*  INP         = (float*)carve((size_t)B_ * L_ * 2 * DI_ * 4);
  float*  XC          = (float*)carve((size_t)B_ * L_ * DI_ * 4);
  float*  DBC         = (float*)carve((size_t)B_ * L_ * 64 * 4);
  float*  DT          = (float*)carve((size_t)B_ * L_ * DI_ * 4);
  float*  U           = (float*)carve((size_t)B_ * L_ * DI_ * 4);
  float*  ERR         = (float*)carve((size_t)B_ * L_ * 4);
  __bf16* ABF         = (__bf16*)carve((size_t)B_ * L_ * DI_ * 2);   // bf16 activations
  __bf16* WBF         = (__bf16*)carve((size_t)2 * DI_ * DM_ * 2);   // bf16 weights

  rank_kernel<<<dim3((L_ + 255) / 256, B_), 256, 0, stream>>>(noise, ids_shuffle, ids_restore, mask);
  {
    int tot = B_ * KEEP_ * DM_;
    embed_kernel<<<(tot + 255) / 256, 256, 0, stream>>>(ecg, ids_shuffle, W_in, b_in, X);
  }

  // convert f32 A (M x lda) and W (N x ldw) to bf16 once, then WMMA GEMM
  auto gemmc = [&](const float* Asrc, int lda, const float* Wsrc, int ldw,
                   float* Cdst, int ldc, int M, int N, int K,
                   const float* bias, int act) {
    long an = (long)M * lda;
    cvt_bf16_kernel<<<(int)((an + 255) / 256), 256, 0, stream>>>(Asrc, ABF, an);
    long wn = (long)N * ldw;
    cvt_bf16_kernel<<<(int)((wn + 255) / 256), 256, 0, stream>>>(Wsrc, WBF, wn);
    dim3 g(N / 64, (M + 31) / 32);
    gemm_wmma_bf16<<<g, 128, 0, stream>>>(ABF, lda, WBF, ldw, Cdst, ldc, M, N, K, bias, act);
  };

  auto mamba = [&](const float* Xin, float* Xout, int Lr,
                   const float* in_w, const float* conv_w, const float* conv_b,
                   const float* xproj_w, const float* dt_w, const float* dt_b,
                   const float* A_log, const float* Dp, const float* out_w) {
    int R = B_ * Lr;
    long tc = (long)R * DI_;
    gemmc(Xin, DM_, in_w, DM_, INP, 2 * DI_, R, 2 * DI_, DM_, nullptr, 0);
    conv_silu_kernel<<<(int)((tc + 255) / 256), 256, 0, stream>>>(INP, conv_w, conv_b, XC, Lr);
    gemmc(XC, DI_, xproj_w, DI_, DBC, 64, R, 64, DI_, nullptr, 0);
    gemmc(DBC, 64, dt_w, DTR_, DT, DI_, R, DI_, DTR_, dt_b, 1);
    scan_kernel<<<B_ * (DI_ / 256), 256, 0, stream>>>(DT, XC, DBC, A_log, Dp, U, Lr);
    gate_kernel<<<(int)((tc + 255) / 256), 256, 0, stream>>>(INP, U, tc);
    gemmc(U, DI_, out_w, DI_, Xout, DM_, R, DM_, DI_, nullptr, 0);
  };

  for (int i = 0; i < 4; ++i) {
    mamba(X, X2, KEEP_,
          enc_in_w   + (size_t)i * 2 * DI_ * DM_,
          enc_conv_w + (size_t)i * DI_ * DC_,
          enc_conv_b + (size_t)i * DI_,
          enc_xproj  + (size_t)i * 64 * DI_,
          enc_dt_w   + (size_t)i * DI_ * DTR_,
          enc_dt_b   + (size_t)i * DI_,
          enc_A_log  + (size_t)i * DI_ * DS_,
          enc_D      + (size_t)i * DI_,
          enc_out_w  + (size_t)i * DM_ * DI_);
    layernorm_kernel<<<B_ * KEEP_, 256, 0, stream>>>(X2, enc_ln_g + (size_t)i * DM_,
                                                     enc_ln_b + (size_t)i * DM_, X);
  }

  {
    long tot = (long)B_ * L_ * DM_;
    build_dec_kernel<<<(int)((tot + 255) / 256), 256, 0, stream>>>(X, ids_restore, mask_token, X2);
  }
  mamba(X2, X, L_, dec_in_w, dec_conv_w, dec_conv_b, dec_xproj,
        dec_dt_w, dec_dt_b, dec_A_log, dec_D, dec_out_w);
  layernorm_kernel<<<B_ * L_, 256, 0, stream>>>(X, dec_ln_g, dec_ln_b, X2);

  float* out   = (float*)d_out;
  float* recon = out + 1;
  head_kernel<<<(B_ * NL_ * L_ + 255) / 256, 256, 0, stream>>>(X2, W_head, b_head, recon);
  err_kernel<<<(B_ * L_ + 255) / 256, 256, 0, stream>>>(recon, ecg, mask, ERR);
  loss_kernel<<<1, 256, 0, stream>>>(ERR, mask, out);
}
